// AnchorAttention_18262200943478
// MI455X (gfx1250) — compile-verified, matrix operand order checked by reading, not run on hardware
//
#include <hip/hip_runtime.h>

typedef unsigned short u16;
typedef unsigned int   u32;
typedef unsigned long long u64;
typedef __attribute__((ext_vector_type(16))) __bf16 v16bf;
typedef __attribute__((ext_vector_type(8)))  float  v8f;

// ---------- constants (from setup_inputs) ----------
#define DD   1024
#define BB   2
#define SS   8192
#define AA   1024
#define HH   16
#define HD   64
#define SQ   7168   // SS - AA

union FragAB { v16bf v; uint4 q[2]; };
union FragC  { v8f  v; float f[8]; };

__device__ __forceinline__ u16 f2bf(float f) {
    u32 u = __builtin_bit_cast(u32, f);
    u32 r = u + 0x7FFFu + ((u >> 16) & 1u);   // round-to-nearest-even
    return (u16)(r >> 16);
}

// ---------- DPP16 row (16-lane group) reductions: xor1,xor2,xor7,xor15 network ----------
template<int CTRL>
__device__ __forceinline__ float dppf(float x) {
    return __builtin_bit_cast(float,
        __builtin_amdgcn_update_dpp(0, __builtin_bit_cast(int, x),
                                    CTRL, 0xF, 0xF, true));
}
__device__ __forceinline__ float rowmax16(float x) {
    x = fmaxf(x, dppf<0xB1>(x));    // quad_perm(1,0,3,2)
    x = fmaxf(x, dppf<0x4E>(x));    // quad_perm(2,3,0,1)
    x = fmaxf(x, dppf<0x141>(x));   // row_half_mirror (xor 7)
    x = fmaxf(x, dppf<0x140>(x));   // row_mirror      (xor 15)
    return x;
}
__device__ __forceinline__ float rowsum16(float x) {
    x += dppf<0xB1>(x);
    x += dppf<0x4E>(x);
    x += dppf<0x141>(x);
    x += dppf<0x140>(x);
    return x;
}

// ---------- async copy: 32 contiguous bytes global -> LDS (two b128, ASYNCcnt) ----------
__device__ __forceinline__ void async_g2l_32B(u32 lds_off, u64 gaddr) {
    asm volatile("global_load_async_to_lds_b128 %0, %1, off"
                 :: "v"(lds_off), "v"(gaddr) : "memory");
    asm volatile("global_load_async_to_lds_b128 %0, %1, off offset:16"
                 :: "v"(lds_off), "v"(gaddr) : "memory");
}
__device__ __forceinline__ void wait_async0() {
    asm volatile("s_wait_asynccnt 0x0" ::: "memory");
}
__device__ __forceinline__ u32 lds_off32(const void* p) {
    // generic pointers into LDS: low 32 bits are the wave-relative LDS offset
    return (u32)(size_t)p;
}

// ---------- fp32 -> bf16 conversion (vectorized) ----------
__global__ __launch_bounds__(256)
void cvt_k(const float* __restrict__ src, u16* __restrict__ dst, int n4) {
    int i = blockIdx.x * blockDim.x + threadIdx.x;
    if (i >= n4) return;
    float4 f = ((const float4*)src)[i];
    u32 lo = ((u32)f2bf(f.y) << 16) | f2bf(f.x);
    u32 hi = ((u32)f2bf(f.w) << 16) | f2bf(f.z);
    ((uint2*)dst)[i] = make_uint2(lo, hi);
}

// ---------- generic bf16 GEMM: C(MxN) = A(MxK=1024) * W(KxN) + bias ----------
// block tile 128x128, 8 waves (4Mx2N), wave tile 32x64, K-step 32.
// A tile staged by async DMA to LDS; B tile repacked (transpose+pair) by VALU.
// MODE 0: QKV epilogue, MODE 1: Q epilogue, MODE 2: proj epilogue (fp32)
template<int MODE>
__global__ __launch_bounds__(256)
void gemm_k(const u16* __restrict__ Ag, const u16* __restrict__ Wg,
            const float* __restrict__ bias,
            u16* __restrict__ q_all, u16* __restrict__ kbuf, u16* __restrict__ vbuf,
            float* __restrict__ outf,
            int N, int rows_per_batch, int batch_stride, int row_offset) {
    const int K = DD;
    __shared__ u32 As[128 * 16];   // [m][kp] pair-packed bf16, 8KB
    __shared__ u32 Bs[128 * 16];   // [n][kp] pair-packed bf16, 8KB

    const int tid   = threadIdx.x;
    const int lane  = tid & 31;
    const int wave  = tid >> 5;
    const int wm    = (wave >> 1) * 32;
    const int wn    = (wave & 1) * 64;
    const int lrow  = lane & 15;
    const int khalf = lane >> 4;
    const int blockM = blockIdx.y * 128;
    const int blockN = blockIdx.x * 128;

    // A staging: thread -> (row, 16-ushort half); straight copy (K pairs contiguous)
    const int a_m    = tid >> 1;
    const int a_half = tid & 1;
    const int r0 = blockM + a_m;
    const int ab = r0 / rows_per_batch;
    const int xrow = ab * batch_stride + row_offset + (r0 - ab * rows_per_batch);
    const u64 abase = (u64)(size_t)(Ag + (size_t)xrow * K) + (u64)(a_half * 32);
    const u32 a_lds = lds_off32(&As[a_m * 16 + a_half * 8]);

    // B staging: thread -> (k pair, 8 n)
    const int b_kp = tid >> 4;          // 0..15
    const int b_n0 = (tid & 15) * 8;    // 0..120

    FragC acc[2][4];
#pragma unroll
    for (int i = 0; i < 2; i++)
#pragma unroll
        for (int j = 0; j < 4; j++)
#pragma unroll
            for (int g = 0; g < 8; g++) acc[i][j].f[g] = 0.f;

    for (int kiter = 0; kiter < K; kiter += 32) {
        __syncthreads();
        // stage A: async DMA, 32B per thread
        async_g2l_32B(a_lds, abase + (u64)(kiter * 2));
        {   // stage B: 32x128 bf16 -> transposed pair-packed [n][kp]
            const u16* w0 = Wg + (size_t)(kiter + 2 * b_kp) * N + blockN + b_n0;
            uint4 q0 = *(const uint4*)w0;
            uint4 q1 = *(const uint4*)(w0 + N);
            const u32* p0 = (const u32*)&q0;
            const u32* p1 = (const u32*)&q1;
#pragma unroll
            for (int j = 0; j < 8; j++) {
                u16 a = (j & 1) ? (u16)(p0[j >> 1] >> 16) : (u16)p0[j >> 1];
                u16 b = (j & 1) ? (u16)(p1[j >> 1] >> 16) : (u16)p1[j >> 1];
                Bs[(b_n0 + j) * 16 + b_kp] = ((u32)b << 16) | a;
            }
        }
        wait_async0();
        __syncthreads();

        const uint4* As4 = (const uint4*)As;
        const uint4* Bs4 = (const uint4*)Bs;
        FragAB aF[2], bF[4];
#pragma unroll
        for (int mi = 0; mi < 2; mi++) {
            int rr = wm + mi * 16 + lrow;
            aF[mi].q[0] = As4[rr * 4 + khalf];
            aF[mi].q[1] = As4[rr * 4 + 2 + khalf];
        }
#pragma unroll
        for (int ni = 0; ni < 4; ni++) {
            int cc = wn + ni * 16 + lrow;
            bF[ni].q[0] = Bs4[cc * 4 + khalf];
            bF[ni].q[1] = Bs4[cc * 4 + 2 + khalf];
        }
#pragma unroll
        for (int mi = 0; mi < 2; mi++)
#pragma unroll
            for (int ni = 0; ni < 4; ni++)
                acc[mi][ni].v = __builtin_amdgcn_wmma_f32_16x16x32_bf16(
                    false, aF[mi].v, false, bF[ni].v,
                    (short)0, acc[mi][ni].v, false, false);
    }

    // epilogue
#pragma unroll
    for (int mi = 0; mi < 2; mi++) {
#pragma unroll
        for (int ni = 0; ni < 4; ni++) {
            int col = blockN + wn + ni * 16 + lrow;
            float bv = bias[col];
#pragma unroll
            for (int g = 0; g < 8; g++) {
                int row = blockM + wm + mi * 16 + g + 8 * khalf;
                float val = acc[mi][ni].f[g] + bv;
                if (MODE == 2) {
                    outf[(size_t)row * N + col] = val;
                } else if (MODE == 1) {
                    int hh = col >> 6, d = col & 63;
                    int b = row / rows_per_batch;
                    int t = row - b * rows_per_batch;
                    q_all[(((size_t)(b * HH + hh)) * SS + AA + t) * HD + d] = f2bf(val);
                } else {
                    int which = col >> 10, cc = col & 1023;
                    int hh = cc >> 6, d = cc & 63;
                    int b = row >> 10, t = row & 1023;
                    size_t hb = (size_t)(b * HH + hh);
                    if (which == 0)      q_all[(hb * SS + t) * HD + d] = f2bf(val);
                    else if (which == 1) kbuf[(hb * AA + t) * HD + d] = f2bf(val);
                    else                 vbuf[(hb * AA + t) * HD + d] = f2bf(val);
                }
            }
        }
    }
}

// ---------- flash attention over anchors: per (b,h), q(S x 64) vs K/V(A x 64) ----------
// grid.x = S/64 (4 waves x 16 rows), grid.y = B*H. 32-key chunks.
// K tile staged by async DMA; V transposed by VALU; row stats via DPP16.
__global__ __launch_bounds__(128)
void attn_k(const u16* __restrict__ q_all, const u16* __restrict__ kb,
            const u16* __restrict__ vb, u16* __restrict__ aout) {
    __shared__ u32 Ks[32 * 32];     // [key][hd-pair]  4KB
    __shared__ u32 Vt[64 * 16];     // [hd][key-pair]  4KB
    __shared__ u32 Ps[4][16 * 16];  // per-wave P tile [row][key-pair] 4KB

    const int tid   = threadIdx.x;
    const int lane  = tid & 31;
    const int wave  = tid >> 5;
    const int lrow  = lane & 15;
    const int khalf = lane >> 4;
    const int bh = blockIdx.y;
    const int b = bh >> 4, h = bh & 15;
    const u16* qb = q_all + (size_t)bh * SS * HD;
    const u16* kg = kb + (size_t)bh * AA * HD;
    const u16* vg = vb + (size_t)bh * AA * HD;
    const int rowbase = blockIdx.x * 64 + wave * 16;

    // q fragments: 16 rows x 64 hd = two K=32 A-fragments, held in regs all loop
    FragAB qf[2];
    {
        const uint4* qr = (const uint4*)(qb + (size_t)(rowbase + lrow) * HD);
#pragma unroll
        for (int kc = 0; kc < 2; kc++) {
            qf[kc].q[0] = qr[kc * 4 + khalf];
            qf[kc].q[1] = qr[kc * 4 + 2 + khalf];
        }
    }

    // K staging (straight copy): 32B per thread
    const u32 k_lds = lds_off32(&Ks[tid * 8]);
    const u64 kbase = (u64)(size_t)kg + (u64)(tid * 32);

    FragC acc[4];
    float m[8], l[8];
#pragma unroll
    for (int g = 0; g < 8; g++) { m[g] = -3.0e38f; l[g] = 0.f; }
#pragma unroll
    for (int t = 0; t < 4; t++)
#pragma unroll
        for (int g = 0; g < 8; g++) acc[t].f[g] = 0.f;

    const float scale = 0.125f;   // 1/sqrt(64)

    for (int kt = 0; kt < AA / 32; kt++) {
        __syncthreads();
        // stage K chunk (32 x 64) via async DMA to LDS
        async_g2l_32B(k_lds, kbase + (u64)(kt * 32 * HD * 2));
        {   // stage V chunk transposed: [hd][key-pair]
            int kp = tid >> 3;
            int d0 = (tid & 7) * 8;
            uint4 q0 = *(const uint4*)(vg + (size_t)(kt * 32 + 2 * kp) * HD + d0);
            uint4 q1 = *(const uint4*)(vg + (size_t)(kt * 32 + 2 * kp + 1) * HD + d0);
            const u32* p0 = (const u32*)&q0;
            const u32* p1 = (const u32*)&q1;
#pragma unroll
            for (int j = 0; j < 8; j++) {
                u16 a = (j & 1) ? (u16)(p0[j >> 1] >> 16) : (u16)p0[j >> 1];
                u16 c = (j & 1) ? (u16)(p1[j >> 1] >> 16) : (u16)p1[j >> 1];
                Vt[(d0 + j) * 16 + kp] = ((u32)c << 16) | a;
            }
        }
        wait_async0();
        __syncthreads();

        // scores: s(16 x 32) = q(16x64) @ k^T
        FragC s[2];
        const uint4* Ks4 = (const uint4*)Ks;
#pragma unroll
        for (int nt = 0; nt < 2; nt++) {
#pragma unroll
            for (int g = 0; g < 8; g++) s[nt].f[g] = 0.f;
            int key = nt * 16 + lrow;
#pragma unroll
            for (int kc = 0; kc < 2; kc++) {
                FragAB kf;
                kf.q[0] = Ks4[key * 8 + kc * 4 + khalf];
                kf.q[1] = Ks4[key * 8 + kc * 4 + 2 + khalf];
                s[nt].v = __builtin_amdgcn_wmma_f32_16x16x32_bf16(
                    false, qf[kc].v, false, kf.v, (short)0, s[nt].v, false, false);
            }
        }

        // online softmax: row stats within 16-lane C-layout groups via DPP16
        u16* Psw = (u16*)Ps[wave];
#pragma unroll
        for (int g = 0; g < 8; g++) {
            float s0 = s[0].f[g] * scale, s1 = s[1].f[g] * scale;
            float rm = rowmax16(fmaxf(s0, s1));
            float mnew = fmaxf(m[g], rm);
            float fac = __expf(m[g] - mnew);
            float p0 = __expf(s0 - mnew);
            float p1 = __expf(s1 - mnew);
            float rs = rowsum16(p0 + p1);
            l[g] = l[g] * fac + rs;
            m[g] = mnew;
#pragma unroll
            for (int t = 0; t < 4; t++) acc[t].f[g] *= fac;
            int prow = g + 8 * khalf;
            Psw[prow * 32 + lrow]      = f2bf(p0);
            Psw[prow * 32 + 16 + lrow] = f2bf(p1);
        }

        // PV: acc(16x64) += P(16x32) @ V(32x64)
        FragAB pf;
        const uint4* Ps4 = (const uint4*)Ps[wave];
        pf.q[0] = Ps4[lrow * 4 + khalf];
        pf.q[1] = Ps4[lrow * 4 + 2 + khalf];
        const uint4* Vt4 = (const uint4*)Vt;
#pragma unroll
        for (int t = 0; t < 4; t++) {
            int d = t * 16 + lrow;
            FragAB vf;
            vf.q[0] = Vt4[d * 4 + khalf];
            vf.q[1] = Vt4[d * 4 + 2 + khalf];
            acc[t].v = __builtin_amdgcn_wmma_f32_16x16x32_bf16(
                false, pf.v, false, vf.v, (short)0, acc[t].v, false, false);
        }
    }

    // normalize and write bf16 (B,S,D) layout for the projection GEMM
#pragma unroll
    for (int t = 0; t < 4; t++) {
#pragma unroll
        for (int g = 0; g < 8; g++) {
            int row = rowbase + g + 8 * khalf;
            int col = h * HD + t * 16 + lrow;
            float o = acc[t].f[g] / l[g];
            aout[((size_t)(b * SS + row)) * DD + col] = f2bf(o);
        }
    }
}

// ---------- host ----------
extern "C" void kernel_launch(void* const* d_in, const int* in_sizes, int n_in,
                              void* d_out, int out_size, void* d_ws, size_t ws_size,
                              hipStream_t stream) {
    const float* x     = (const float*)d_in[0];
    const float* Wqkv  = (const float*)d_in[1];
    const float* bqkv  = (const float*)d_in[2];
    const float* Wq    = (const float*)d_in[3];
    const float* bq    = (const float*)d_in[4];
    const float* Wproj = (const float*)d_in[5];
    const float* bproj = (const float*)d_in[6];
    float* out = (float*)d_out;

    char* ws = (char*)d_ws;
    size_t off = 0;
    u16* xb     = (u16*)(ws + off); off += (size_t)BB * SS * DD * 2;       // 32 MB
    u16* wqkvb  = (u16*)(ws + off); off += (size_t)DD * 3 * DD * 2;        // 6 MB
    u16* wqb    = (u16*)(ws + off); off += (size_t)DD * DD * 2;            // 2 MB
    u16* wprojb = (u16*)(ws + off); off += (size_t)DD * DD * 2;            // 2 MB
    u16* q_all  = (u16*)(ws + off); off += (size_t)BB * HH * SS * HD * 2;  // 32 MB
    u16* kbuf   = (u16*)(ws + off); off += (size_t)BB * HH * AA * HD * 2;  // 4 MB
    u16* vbuf   = (u16*)(ws + off); off += (size_t)BB * HH * AA * HD * 2;  // 4 MB
    u16* aoutb  = (u16*)(ws + off); off += (size_t)BB * SS * DD * 2;       // 32 MB

    // fp32 -> bf16
    {
        int n4 = (BB * SS * DD) / 4;
        cvt_k<<<(n4 + 255) / 256, 256, 0, stream>>>(x, xb, n4);
        n4 = (DD * 3 * DD) / 4;
        cvt_k<<<(n4 + 255) / 256, 256, 0, stream>>>(Wqkv, wqkvb, n4);
        n4 = (DD * DD) / 4;
        cvt_k<<<(n4 + 255) / 256, 256, 0, stream>>>(Wq, wqb, n4);
        cvt_k<<<(n4 + 255) / 256, 256, 0, stream>>>(Wproj, wprojb, n4);
    }

    // anchors @ Wqkv : M = B*A = 2048, N = 3072
    {
        dim3 g(3 * DD / 128, (BB * AA) / 128);
        gemm_k<0><<<g, 256, 0, stream>>>(xb, wqkvb, bqkv, q_all, kbuf, vbuf,
                                         nullptr, 3 * DD, AA, SS, 0);
    }
    // queries @ Wq : M = B*Sq = 14336, N = 1024
    {
        dim3 g(DD / 128, (BB * SQ) / 128);
        gemm_k<1><<<g, 256, 0, stream>>>(xb, wqb, bq, q_all, nullptr, nullptr,
                                         nullptr, DD, SQ, SS, AA);
    }
    // attention
    {
        dim3 g(SS / 64, BB * HH);
        attn_k<<<g, 128, 0, stream>>>(q_all, kbuf, vbuf, aoutb);
    }
    // attn_out @ Wproj : M = B*S = 16384, N = 1024, fp32 out
    {
        dim3 g(DD / 128, (BB * SS) / 128);
        gemm_k<2><<<g, 256, 0, stream>>>(aoutb, wprojb, bproj, nullptr, nullptr,
                                         nullptr, out, DD, BB * SS, 0, 0);
    }
}
